// VQModule_77687368450621
// MI455X (gfx1250) — compile-verified, matrix operand order checked by reading, not run on hardware
//
#include <hip/hip_runtime.h>

// Problem constants (reference shapes)
#define BDIM  32
#define CDIM  256
#define HDIM  32
#define WDIM  32
#define KDICT 8192
#define NROWS (BDIM * HDIM * WDIM)   // 32768 flattened vectors

// Tiling
#define MTILE 128                     // rows per block (8 waves x 16)
#define NTILE 64                      // codebook columns per LDS tile
#define NTILES (KDICT / NTILE)        // 128 tiles
#define KC_CHUNKS (CDIM / 32)         // 8 k-chunks of 32
#define RB (MTILE / 16)               // 8 row fragments
#define NB (NTILE / 16)               // 4 col fragments

// LDS layout (bytes): A frags [0,128K), B buf0 [128K,192K), B buf1 [192K,256K)
#define A_LDS_BYTES   (2 * KC_CHUNKS * RB * 32 * 16 * 2)   // 131072
#define B_TILE_BYTES  (2 * KC_CHUNKS * NB * 32 * 16 * 2)   // 65536
#define TOTAL_LDS     (A_LDS_BYTES + 2 * B_TILE_BYTES)     // 262144

typedef __attribute__((ext_vector_type(16))) __bf16       v16bf;
typedef __attribute__((ext_vector_type(8)))  float        v8f;
typedef __attribute__((ext_vector_type(4)))  unsigned int v4u;

struct frag_bits { v4u lo, hi; };     // 32 bytes == one lane's 16 bf16 values

static __device__ inline unsigned short f2bf(float f) {
  // round-to-nearest-even f32 -> bf16
  unsigned u = __builtin_bit_cast(unsigned, f);
  unsigned r = u + 0x7FFFu + ((u >> 16) & 1u);
  return (unsigned short)(r >> 16);
}
static __device__ inline float bf2f(unsigned short s) {
  unsigned u = ((unsigned)s) << 16;
  return __builtin_bit_cast(float, u);
}
static __device__ inline v16bf ld_frag(const v4u* p) {
  frag_bits t;
  t.lo = p[0];
  t.hi = p[1];
  return __builtin_bit_cast(v16bf, t);
}

// ---- CDNA5 async global->LDS copy (ASYNCcnt-tracked), inline asm path -----
// GVS addressing: mem = SGPR64 + VGPR32 + IOFFSET ; VDST vgpr = LDS byte addr.
static __device__ inline void async_g2l_b128(unsigned lds_addr, unsigned gl_off,
                                             const void* sbase) {
  asm volatile("global_load_async_to_lds_b128 %0, %1, %2 offset:0"
               :
               : "v"(lds_addr), "v"(gl_off), "s"(sbase)
               : "memory");
}
static __device__ inline void wait_async0() {
  asm volatile("s_wait_asynccnt 0" ::: "memory");
}

// fragment index helpers (ushort granularity)
#define B_FRAG_US(var, kc, nb, fl, slot) \
  (((((var) * KC_CHUNKS + (kc)) * NB + (nb)) * 32 + (fl)) * 16 + (slot))

// ---------------------------------------------------------------------------
// Kernel 1: e2[k] = sum_c embed[c,k]^2
// ---------------------------------------------------------------------------
__global__ __launch_bounds__(256)
void vq_e2_kernel(const float* __restrict__ embed, float* __restrict__ e2) {
  int k = blockIdx.x * 256 + threadIdx.x;
  float s = 0.0f;
  #pragma unroll 8
  for (int c = 0; c < CDIM; ++c) {
    float v = embed[(size_t)c * KDICT + k];
    s += v * v;
  }
  e2[k] = s;
}

// ---------------------------------------------------------------------------
// Kernel 2: one-time convert of embed into bf16 hi/lo WMMA B-fragments,
// tile-major in workspace. Chunk nt (64KB) is byte-identical to the LDS
// B-tile image, so the GEMM kernel can stream it with raw async copies.
// ---------------------------------------------------------------------------
__global__ __launch_bounds__(256)
void vq_prep_b_kernel(const float* __restrict__ embed,
                      unsigned short* __restrict__ bfrag) {
  const int nt  = blockIdx.x;                 // 0..127
  unsigned short* chunk = bfrag + (size_t)nt * (B_TILE_BYTES / 2);
  for (int e = threadIdx.x; e < CDIM * NTILE; e += 256) {
    int c  = e >> 6;                          // / NTILE
    int nn = e & (NTILE - 1);
    float ev = embed[(size_t)c * KDICT + nt * NTILE + nn];
    unsigned short hb = f2bf(ev);
    unsigned short lb = f2bf(ev - bf2f(hb));
    int kc = c >> 5;
    int k  = c & 31;
    int nb = nn >> 4;
    int fl   = (nn & 15) + (((k & 15) >= 8) ? 16 : 0);
    int slot = (k & 7) + ((k >= 16) ? 8 : 0);
    chunk[B_FRAG_US(0, kc, nb, fl, slot)] = hb;
    chunk[B_FRAG_US(1, kc, nb, fl, slot)] = lb;
  }
}

// ---------------------------------------------------------------------------
// Kernel 3: split-bf16 WMMA GEMM + per-row argmin over K=8192.
// A resident in LDS; B tiles double-buffered via global_load_async_to_lds.
// ---------------------------------------------------------------------------
__global__ __launch_bounds__(256)
void vq_dist_argmin_kernel(const float* __restrict__ x,
                           const unsigned short* __restrict__ bfrag,
                           const float* __restrict__ e2,
                           int* __restrict__ ids) {
  extern __shared__ unsigned short smem_us[];
  v4u* smem_v4 = (v4u*)smem_us;

  const int tid  = threadIdx.x;
  const int lane = tid & 31;
  const int wv   = tid >> 5;              // wave id 0..7 -> M fragment row
  const int n0   = blockIdx.x * MTILE;    // first flattened row of this block

  const unsigned lds_base = (unsigned)(uintptr_t)(&smem_us[0]); // LDS byte offset

  // ---- issue async fetch of B tile 0 into buffer 0 (overlaps A convert) ----
  {
    unsigned off = (unsigned)tid * 16u;
    #pragma unroll
    for (int i = 0; i < B_TILE_BYTES / (256 * 16); ++i) {  // 16 iters
      async_g2l_b128(lds_base + A_LDS_BYTES + off, off, (const void*)bfrag);
      off += 256 * 16;
    }
  }

  // ---- Load A: rows n0..n0+127, all 256 channels, split into bf16 hi/lo ----
  for (int e = tid; e < MTILE * CDIM; e += 256) {
    int row = e >> 8;                     // / CDIM
    int c   = e & (CDIM - 1);
    int n   = n0 + row;
    int b   = n >> 10;                    // / (H*W)
    int w   = (n >> 5) & 31;
    int h   = n & 31;
    // flatten[n, c] = x[b, c, h, w]
    float xv = x[(((size_t)b * CDIM + c) * HDIM + h) * WDIM + w];
    unsigned short hb = f2bf(xv);
    unsigned short lb = f2bf(xv - bf2f(hb));
    int kc = c >> 5;
    int k  = c & 31;
    int rb = row >> 4;
    int fl   = (row & 15) + (((k & 15) >= 8) ? 16 : 0);
    int slot = (k & 7) + ((k >= 16) ? 8 : 0);
    smem_us[(((0 * KC_CHUNKS + kc) * RB + rb) * 32 + fl) * 16 + slot] = hb;
    smem_us[(((1 * KC_CHUNKS + kc) * RB + rb) * 32 + fl) * 16 + slot] = lb;
  }
  wait_async0();
  __syncthreads();

  float bestv[8];
  int   besti[8];
  #pragma unroll
  for (int r = 0; r < 8; ++r) { bestv[r] = 3.4e38f; besti[r] = 0; }

  const v8f vzero = {0.f, 0.f, 0.f, 0.f, 0.f, 0.f, 0.f, 0.f};

  for (int nt = 0; nt < NTILES; ++nt) {
    const int cur = nt & 1;
    // ---- prefetch next tile into the other buffer (written last iter-1,
    //      all readers released by the trailing barrier) ----
    if (nt + 1 < NTILES) {
      unsigned gbase = (unsigned)(nt + 1) * B_TILE_BYTES;
      unsigned lbase = lds_base + A_LDS_BYTES + (unsigned)((nt + 1) & 1) * B_TILE_BYTES;
      unsigned off = (unsigned)tid * 16u;
      #pragma unroll
      for (int i = 0; i < B_TILE_BYTES / (256 * 16); ++i) {
        async_g2l_b128(lbase + off, gbase + off, (const void*)bfrag);
        off += 256 * 16;
      }
    }

    // ---- GEMM on current buffer: 3 WMMAs per (kc, nb) for f32-via-bf16 ----
    const int bv4 = (A_LDS_BYTES + cur * B_TILE_BYTES) / 16;  // v4u index base
    v8f acc[NB];
    #pragma unroll
    for (int nb = 0; nb < NB; ++nb) acc[nb] = vzero;

    #pragma unroll
    for (int kc = 0; kc < KC_CHUNKS; ++kc) {
      v16bf a_hi = ld_frag(&smem_v4[(((0 * KC_CHUNKS + kc) * RB + wv) * 32 + lane) * 2]);
      v16bf a_lo = ld_frag(&smem_v4[(((1 * KC_CHUNKS + kc) * RB + wv) * 32 + lane) * 2]);
      #pragma unroll
      for (int nb = 0; nb < NB; ++nb) {
        v16bf b_hi = ld_frag(&smem_v4[bv4 + (((0 * KC_CHUNKS + kc) * NB + nb) * 32 + lane) * 2]);
        v16bf b_lo = ld_frag(&smem_v4[bv4 + (((1 * KC_CHUNKS + kc) * NB + nb) * 32 + lane) * 2]);
        acc[nb] = __builtin_amdgcn_wmma_f32_16x16x32_bf16(
            false, a_hi, false, b_hi, (short)0, acc[nb], false, false);
        acc[nb] = __builtin_amdgcn_wmma_f32_16x16x32_bf16(
            false, a_hi, false, b_lo, (short)0, acc[nb], false, false);
        acc[nb] = __builtin_amdgcn_wmma_f32_16x16x32_bf16(
            false, a_lo, false, b_hi, (short)0, acc[nb], false, false);
      }
    }

    // ---- dist = ||e||^2 - 2*(z.e) ; running argmin per (row, lane-col) ----
    #pragma unroll
    for (int nb = 0; nb < NB; ++nb) {
      int ncol = nt * NTILE + nb * 16 + (lane & 15);
      float e2v = e2[ncol];
      #pragma unroll
      for (int r = 0; r < 8; ++r) {
        float d = e2v - 2.0f * acc[nb][r];
        if (d < bestv[r]) { bestv[r] = d; besti[r] = ncol; }
      }
    }

    wait_async0();     // next tile landed in LDS
    __syncthreads();   // everyone done reading current buffer
  }

  // ---- reduce argmin across the 16 lanes holding the same row ----
  #pragma unroll
  for (int r = 0; r < 8; ++r) {
    float bv = bestv[r];
    int   bi = besti[r];
    #pragma unroll
    for (int m = 8; m >= 1; m >>= 1) {
      float ov = __shfl_xor(bv, m, 16);
      int   oi = __shfl_xor(bi, m, 16);
      if (ov < bv || (ov == bv && oi < bi)) { bv = ov; bi = oi; }
    }
    if ((lane & 15) == 0) {
      // lanes 0..15 hold rows r, lanes 16..31 hold rows r+8 (C/D layout)
      int row = wv * 16 + r + ((lane & 16) ? 8 : 0);
      ids[n0 + row] = bi;
    }
  }
}

// ---------------------------------------------------------------------------
// Kernel 4: gather. out[b,c,h,w] = embed[c, ids[b, w, h]]; coalesced over w.
// ---------------------------------------------------------------------------
__global__ __launch_bounds__(256)
void vq_gather_kernel(const float* __restrict__ embed,
                      const int* __restrict__ ids,
                      float* __restrict__ out) {
  int bid = blockIdx.x;         // 0..B*H-1
  int b = bid >> 5;
  int h = bid & 31;
  int tx = threadIdx.x & 31;    // w
  int ty = threadIdx.x >> 5;    // c phase 0..7
  int id = ids[b * (HDIM * WDIM) + tx * 32 + h];
  for (int c = ty; c < CDIM; c += 8) {
    out[(((size_t)b * CDIM + c) * HDIM + h) * WDIM + tx] =
        embed[(size_t)c * KDICT + id];
  }
}

// ---------------------------------------------------------------------------
extern "C" void kernel_launch(void* const* d_in, const int* in_sizes, int n_in,
                              void* d_out, int out_size, void* d_ws, size_t ws_size,
                              hipStream_t stream) {
  const float* x     = (const float*)d_in[0];
  const float* embed = (const float*)d_in[1];
  float* out = (float*)d_out;
  int*   ids = (int*)d_out + (size_t)BDIM * CDIM * HDIM * WDIM;  // 2nd output

  // workspace layout: [B fragments: 8MB][e2: 32KB]
  unsigned short* bfrag = (unsigned short*)d_ws;
  float* e2 = (float*)((char*)d_ws + (size_t)NTILES * B_TILE_BYTES);

  vq_e2_kernel<<<KDICT / 256, 256, 0, stream>>>(embed, e2);
  vq_prep_b_kernel<<<NTILES, 256, 0, stream>>>(embed, bfrag);
  vq_dist_argmin_kernel<<<NROWS / MTILE, 256, TOTAL_LDS, stream>>>(x, bfrag, e2, ids);
  vq_gather_kernel<<<BDIM * HDIM, 256, 0, stream>>>(embed, ids, out);
}
